// DGCN2_14370960572499
// MI455X (gfx1250) — matrix operand order repaired and checked by reading, not compiled, not run on hardware
//
#include <hip/hip_runtime.h>
#include <math.h>

// ---------- problem constants (from reference) ----------
#define NN      102400      // B*T*NODES total node rows
#define DD      64
#define EE      1638400
#define L1      96
#define L2      64
#define GDIM    256         // 4*H LSTM gates
#define HDIM    64
#define TT      32
#define NODES   400
#define BB      8
#define RS      3200        // B*NODES sequence rows
#define F1DIM   128
#define NCLS    3

typedef __attribute__((ext_vector_type(16))) __bf16 v16bf;
typedef __attribute__((ext_vector_type(8)))  __bf16 v8bf;
typedef __attribute__((ext_vector_type(8)))  float  v8f;

__device__ inline float sigf(float x) { return 1.0f / (1.0f + expf(-x)); }

// ---- WMMA fragment loaders (CDNA5 ISA 7.12.2 layouts, wave32) ----
// 16-bit A matrix 16x32 (MxK): lanes 0-15 -> M=lane, lanes 16-31 -> M=lane-16.
// lane's elements are two contiguous 8-element runs: k0+8*half and k0+16+8*half.
__device__ inline v16bf load_a_bf16(const __bf16* __restrict__ A, int lda, int row0, int k0,
                                    int lane) {
    int half = lane >> 4, m = (lane & 15) + row0;
    const __bf16* p = A + (size_t)m * lda + k0 + 8 * half;
    v8bf lo = *(const v8bf*)p;          // 16-byte load
    v8bf hi = *(const v8bf*)(p + 16);   // 16-byte load
    return __builtin_shufflevector(lo, hi, 0, 1, 2, 3, 4, 5, 6, 7,
                                   8, 9, 10, 11, 12, 13, 14, 15);
}
__device__ inline v16bf load_a_f32(const float* __restrict__ A, int lda, int row0, int k0,
                                   int lane) {
    int half = lane >> 4, m = (lane & 15) + row0;
    v16bf a;
#pragma unroll
    for (int e = 0; e < 16; ++e) {
        int k = e + ((e < 8) ? 0 : 8) + 8 * half;
        a[e] = (__bf16)A[(size_t)m * lda + k0 + k];
    }
    return a;
}
// packed-B: fragment-major layout  [tn][kb][lane][16 elems]  (32 contiguous bytes per lane)
__device__ inline v16bf load_b_packed(const __bf16* __restrict__ Bp, int kblocks, int tn,
                                      int kb, int lane) {
    return *(const v16bf*)(Bp + ((size_t)(tn * kblocks + kb) * 32 + lane) * 16);
}

// ---------- column statistics (mean + 1/std, ddof=1) ----------
__global__ void colstats_kernel(const float* __restrict__ x, float* mean, float* rstd, int n, int d) {
    int f = blockIdx.x, tid = threadIdx.x;
    float s = 0.f, sq = 0.f;
    for (int i = tid; i < n; i += blockDim.x) {
        float v = x[(size_t)i * d + f];
        s += v; sq += v * v;
    }
    __shared__ float ss[256], sqq[256];
    ss[tid] = s; sqq[tid] = sq; __syncthreads();
    for (int o = 128; o > 0; o >>= 1) {
        if (tid < o) { ss[tid] += ss[tid + o]; sqq[tid] += sqq[tid + o]; }
        __syncthreads();
    }
    if (tid == 0) {
        float m = ss[0] / (float)n;
        float var = (sqq[0] - (float)n * m * m) / (float)(n - 1);
        mean[f] = m;
        rstd[f] = rsqrtf(var);
    }
}

__global__ void norm_bf16_kernel(const float* __restrict__ x, const float* mean,
                                 const float* rstd, __bf16* out, int n, int d) {
    int i = blockIdx.x * blockDim.x + threadIdx.x;
    if (i >= n * d) return;
    int f = i % d;
    out[i] = (__bf16)((x[i] - mean[f]) * rstd[f]);
}

// ---------- weight prep: write B operand directly in WMMA fragment-major layout ----------
// B[k][n] = transpose ? src[n*K + k] : src[k*N + n]
__global__ void pack_b_kernel(const float* __restrict__ src, __bf16* __restrict__ dst,
                              int K, int N, int transpose) {
    int i = blockIdx.x * blockDim.x + threadIdx.x;
    if (i >= K * N) return;
    int kblocks = K >> 5;
    int e = i & 15;
    int lane = (i >> 4) & 31;
    int blk = i >> 9;
    int kb = blk % kblocks;
    int tn = blk / kblocks;
    int k = kb * 32 + (e & 7) + 8 * (lane >> 4) + 16 * (e >> 3);
    int n = tn * 16 + (lane & 15);
    float v = transpose ? src[(size_t)n * K + k] : src[(size_t)k * N + n];
    dst[i] = (__bf16)v;
}
__global__ void gate_bias_kernel(const float* bi, const float* bh, float* gb) {
    int i = threadIdx.x;
    gb[i] = bi[i] + bh[i];
}

// ---------- degree / dinv ----------
__global__ void fill_kernel(float* p, float v, int n) {
    int i = blockIdx.x * blockDim.x + threadIdx.x;
    if (i < n) p[i] = v;
}
__global__ void deg_accum_kernel(const int* __restrict__ dst, const float* __restrict__ ea,
                                 float* deg, int e) {
    int i = blockIdx.x * blockDim.x + threadIdx.x;
    if (i < e) atomicAdd(&deg[dst[i]], fabsf(ea[2 * i]));
}
__global__ void rsqrt_inplace_kernel(float* p, int n) {
    int i = blockIdx.x * blockDim.x + threadIdx.x;
    if (i >= n) return;
    float v = p[i];
    p[i] = (v > 0.f) ? rsqrtf(v) : 0.f;
}

// ---------- generic WMMA bf16 GEMM: C[MxN] = A[MxK] @ B[KxN] (+bias) (opt relu) ----------
// grid = tiles/4 blocks, 128 threads = 4 waves, one 16x16 C tile per wave
__global__ void __launch_bounds__(128)
wmma_gemm_kernel(const __bf16* __restrict__ A, const __bf16* __restrict__ Bp,
                 float* __restrict__ C, const float* __restrict__ bias,
                 int M, int N, int K, int relu) {
    int ntn = N >> 4;
    int tile = blockIdx.x * 4 + (threadIdx.x >> 5);
    int tm = tile / ntn, tn = tile % ntn;
    int lane = threadIdx.x & 31;
    int kblocks = K >> 5;
    v8f acc = {};
    for (int kb = 0; kb < kblocks; ++kb) {
        v16bf a = load_a_bf16(A, K, tm * 16, kb * 32, lane);
        v16bf b = load_b_packed(Bp, kblocks, tn, kb, lane);
        acc = __builtin_amdgcn_wmma_f32_16x16x32_bf16(false, a, false, b, (short)0, acc,
                                                      false, false);
    }
    int n = tn * 16 + (lane & 15);
    int half = lane >> 4;
    float bv = bias ? bias[n] : 0.f;
#pragma unroll
    for (int v = 0; v < 8; ++v) {
        int row = tm * 16 + v + 8 * half;
        float x = acc[v] + bv;
        if (relu) x = fmaxf(x, 0.f);
        C[(size_t)row * N + n] = x;
    }
}

// ---------- edge scatter: acc[dst] += h[src] * dinv[src]*|ea|*dinv[dst] ----------
// one wave per edge, lanes stride the feature dim
__global__ void scatter_kernel(const float* __restrict__ h, float* __restrict__ acc,
                               const int* __restrict__ src, const int* __restrict__ dst,
                               const float* __restrict__ ea, const float* __restrict__ dinv,
                               int e, int F) {
    int wid = (blockIdx.x * blockDim.x + threadIdx.x) >> 5;
    int lane = threadIdx.x & 31;
    if (wid >= e) return;
    int s = src[wid], d = dst[wid];
    __builtin_prefetch(&h[(size_t)s * F], 0, 1);   // global_prefetch_b8: random gather row
    float norm = dinv[s] * fabsf(ea[2 * wid]) * dinv[d];
    for (int f = lane; f < F; f += 32)
        atomicAdd(&acc[(size_t)d * F + f], h[(size_t)s * F + f] * norm);
}

// ---------- GCN finalize: self-loop + bias + relu, write f32 (in place) + bf16 ----------
__global__ void gcn_finalize_kernel(float* __restrict__ acc, const float* __restrict__ h,
                                    const float* __restrict__ dinv, const float* __restrict__ bias,
                                    __bf16* __restrict__ bfout, int n, int F) {
    int i = blockIdx.x * blockDim.x + threadIdx.x;
    if (i >= n * F) return;
    int r = i / F, f = i % F;
    float di = dinv[r];
    float v = acc[i] + h[i] * di * di + bias[f];
    v = fmaxf(v, 0.f);
    acc[i] = v;
    bfout[i] = (__bf16)v;
}

// ---------- persistent LSTM: 200 blocks x 16 rows, h/c/W_hh^T (packed) in LDS ----------
__global__ void __launch_bounds__(256)
lstm_kernel(const float* __restrict__ gin, const __bf16* __restrict__ whhP,
            __bf16* __restrict__ houtbf) {
    __shared__ __align__(32) __bf16 s_whh[HDIM * GDIM];  // 32 KB, packed fragment-major
    __shared__ float s_h[16 * HDIM];                     // 4 KB
    __shared__ float s_c[16 * HDIM];                     // 4 KB
    __shared__ float s_g[16 * GDIM];                     // 16 KB
    int tid = threadIdx.x, lane = tid & 31, wid = tid >> 5;
    int rowbase = blockIdx.x * 16;

    for (int i = tid; i < HDIM * GDIM; i += 256) s_whh[i] = whhP[i];
    for (int i = tid; i < 16 * HDIM; i += 256) { s_h[i] = 0.f; s_c[i] = 0.f; }
    __syncthreads();

    for (int t = 0; t < TT; ++t) {
        // gate pre-activations: each wave owns two 16-wide N tiles of the 256 gate cols
#pragma unroll
        for (int sub = 0; sub < 2; ++sub) {
            int tn = wid * 2 + sub;
            int col0 = tn * 16;
            v8f acc = {};
#pragma unroll
            for (int kb = 0; kb < HDIM / 32; ++kb) {
                v16bf a = load_a_f32(s_h, HDIM, 0, kb * 32, lane);
                v16bf b = load_b_packed(s_whh, HDIM / 32, tn, kb, lane);
                acc = __builtin_amdgcn_wmma_f32_16x16x32_bf16(false, a, false, b, (short)0,
                                                              acc, false, false);
            }
            int nloc = lane & 15, half = lane >> 4;
#pragma unroll
            for (int v = 0; v < 8; ++v) {
                int lr = v + 8 * half;
                int r = rowbase + lr;          // sequence row = b*NODES + node
                int bb = r / NODES, node = r - bb * NODES;
                int idx = (bb * TT + t) * NODES + node;   // original node-row index
                s_g[lr * GDIM + col0 + nloc] =
                    acc[v] + gin[(size_t)idx * GDIM + col0 + nloc];
            }
        }
        __syncthreads();
        // gate nonlinearity + state update (i,f,g,o torch order)
        for (int e = tid; e < 16 * HDIM; e += 256) {
            int lr = e >> 6, j = e & 63;
            const float* g = &s_g[lr * GDIM];
            float gi = g[j], gf = g[HDIM + j], gg = g[2 * HDIM + j], go = g[3 * HDIM + j];
            float c = sigf(gf) * s_c[e] + sigf(gi) * tanhf(gg);
            float h = sigf(go) * tanhf(c);
            s_c[e] = c;
            s_h[e] = h;
        }
        __syncthreads();
    }
    for (int e = tid; e < 16 * HDIM; e += 256) {
        int lr = e >> 6, j = e & 63;
        houtbf[(size_t)(rowbase + lr) * HDIM + j] = (__bf16)s_h[e];
    }
}

// ---------- FC2 (128->3) + softmax ----------
__global__ void fc2_softmax_kernel(const float* __restrict__ f1, const float* __restrict__ w,
                                   const float* __restrict__ b, float* __restrict__ out, int rows) {
    int r = blockIdx.x * blockDim.x + threadIdx.x;
    if (r >= rows) return;
    float l[NCLS];
#pragma unroll
    for (int c = 0; c < NCLS; ++c) {
        float s = b[c];
        for (int k = 0; k < F1DIM; ++k) s += f1[(size_t)r * F1DIM + k] * w[c * F1DIM + k];
        l[c] = s;
    }
    float mx = fmaxf(l[0], fmaxf(l[1], l[2]));
    float e0 = expf(l[0] - mx), e1 = expf(l[1] - mx), e2 = expf(l[2] - mx);
    float inv = 1.f / (e0 + e1 + e2);
    out[r * 3 + 0] = e0 * inv;
    out[r * 3 + 1] = e1 * inv;
    out[r * 3 + 2] = e2 * inv;
}

extern "C" void kernel_launch(void* const* d_in, const int* in_sizes, int n_in,
                              void* d_out, int out_size, void* d_ws, size_t ws_size,
                              hipStream_t stream) {
    const float* x        = (const float*)d_in[0];
    const float* ea       = (const float*)d_in[1];
    const float* conv1_w  = (const float*)d_in[2];
    const float* conv1_b  = (const float*)d_in[3];
    const float* conv2_w  = (const float*)d_in[4];
    const float* conv2_b  = (const float*)d_in[5];
    const float* w_ih     = (const float*)d_in[6];
    const float* w_hh     = (const float*)d_in[7];
    const float* b_ih     = (const float*)d_in[8];
    const float* b_hh     = (const float*)d_in[9];
    const float* fc1_w    = (const float*)d_in[10];
    const float* fc1_b    = (const float*)d_in[11];
    const float* fc2_w    = (const float*)d_in[12];
    const float* fc2_b    = (const float*)d_in[13];
    const int*   ei       = (const int*)d_in[14];
    const int*   src      = ei;
    const int*   dst      = ei + EE;
    float* out = (float*)d_out;

    // workspace layout (scratch is NOT zeroed between calls -> zero what we accumulate into)
    char* base = (char*)d_ws;
    size_t off = 0;
    auto alc = [&](size_t bytes) -> void* {
        void* p = base + off;
        off = (off + bytes + 255) & ~(size_t)255;
        return p;
    };
    float*  mean  = (float*)alc(64 * 4);
    float*  rstd  = (float*)alc(64 * 4);
    float*  dinv  = (float*)alc((size_t)NN * 4);
    float*  gbias = (float*)alc(GDIM * 4);
    __bf16* w1p   = (__bf16*)alc((size_t)DD * L1 * 2);      // packed B operands
    __bf16* w2p   = (__bf16*)alc((size_t)L1 * L2 * 2);
    __bf16* wihP  = (__bf16*)alc((size_t)DD * GDIM * 2);
    __bf16* whhP  = (__bf16*)alc((size_t)HDIM * GDIM * 2);
    __bf16* fc1P  = (__bf16*)alc((size_t)L2 * F1DIM * 2);
    __bf16* xbf   = (__bf16*)alc((size_t)NN * DD * 2);
    float*  h1    = (float*)alc((size_t)NN * L1 * 4);   // reused as h2 (NN*L2)
    float*  acc1  = (float*)alc((size_t)NN * L1 * 4);   // reused as acc2
    __bf16* bf1   = (__bf16*)alc((size_t)NN * L1 * 2);  // reused as bf2
    float*  gin   = (float*)alc((size_t)NN * GDIM * 4);
    __bf16* hobf  = (__bf16*)alc((size_t)RS * HDIM * 2);
    float*  f1    = (float*)alc((size_t)RS * F1DIM * 4);

    const int TB = 256;
    auto g1d = [](int n, int tb) { return (n + tb - 1) / tb; };

    // 1. normalize
    colstats_kernel<<<DD, TB, 0, stream>>>(x, mean, rstd, NN, DD);
    norm_bf16_kernel<<<g1d(NN * DD, TB), TB, 0, stream>>>(x, mean, rstd, xbf, NN, DD);

    // 2. weight prep: pack B operands into WMMA fragment-major layout
    pack_b_kernel<<<g1d(DD * L1, TB), TB, 0, stream>>>(conv1_w, w1p, DD, L1, 0);
    pack_b_kernel<<<g1d(L1 * L2, TB), TB, 0, stream>>>(conv2_w, w2p, L1, L2, 0);
    pack_b_kernel<<<g1d(DD * GDIM, TB), TB, 0, stream>>>(w_ih, wihP, DD, GDIM, 1);
    pack_b_kernel<<<g1d(HDIM * GDIM, TB), TB, 0, stream>>>(w_hh, whhP, HDIM, GDIM, 1);
    pack_b_kernel<<<g1d(L2 * F1DIM, TB), TB, 0, stream>>>(fc1_w, fc1P, L2, F1DIM, 1);
    gate_bias_kernel<<<1, GDIM, 0, stream>>>(b_ih, b_hh, gbias);

    // 3. degree -> deg^{-1/2} (self-loop weight 1)
    fill_kernel<<<g1d(NN, TB), TB, 0, stream>>>(dinv, 1.0f, NN);
    deg_accum_kernel<<<g1d(EE, TB), TB, 0, stream>>>(dst, ea, dinv, EE);
    rsqrt_inplace_kernel<<<g1d(NN, TB), TB, 0, stream>>>(dinv, NN);

    // 4. GCN layer 1: GEMM (WMMA) -> scatter -> finalize(relu)
    wmma_gemm_kernel<<<(NN / 16) * (L1 / 16) / 4, 128, 0, stream>>>(xbf, w1p, h1, nullptr,
                                                                    NN, L1, DD, 0);
    fill_kernel<<<g1d(NN * L1, TB), TB, 0, stream>>>(acc1, 0.0f, NN * L1);
    scatter_kernel<<<g1d(EE * 32, TB), TB, 0, stream>>>(h1, acc1, src, dst, ea, dinv, EE, L1);
    gcn_finalize_kernel<<<g1d(NN * L1, TB), TB, 0, stream>>>(acc1, h1, dinv, conv1_b, bf1, NN, L1);

    // 5. GCN layer 2 (reuse h1/acc1/bf1 buffers)
    wmma_gemm_kernel<<<(NN / 16) * (L2 / 16) / 4, 128, 0, stream>>>(bf1, w2p, h1, nullptr,
                                                                    NN, L2, L1, 0);
    fill_kernel<<<g1d(NN * L2, TB), TB, 0, stream>>>(acc1, 0.0f, NN * L2);
    scatter_kernel<<<g1d(EE * 32, TB), TB, 0, stream>>>(h1, acc1, src, dst, ea, dinv, EE, L2);
    gcn_finalize_kernel<<<g1d(NN * L2, TB), TB, 0, stream>>>(acc1, h1, dinv, conv2_b, bf1, NN, L2);

    // 6. LSTM input GEMM: gin = g2 @ w_ih^T + (b_ih + b_hh)   [NN x 256]
    wmma_gemm_kernel<<<(NN / 16) * (GDIM / 16) / 4, 128, 0, stream>>>(bf1, wihP, gin, gbias,
                                                                      NN, GDIM, DD, 0);

    // 7. recurrent LSTM (persistent, LDS-resident state + packed W_hh, WMMA per step)
    lstm_kernel<<<RS / 16, 256, 0, stream>>>(gin, whhP, hobf);

    // 8. FC1 (relu) via WMMA, FC2 + softmax scalar
    wmma_gemm_kernel<<<(RS / 16) * (F1DIM / 16) / 4, 128, 0, stream>>>(hobf, fc1P, f1, fc1_b,
                                                                       RS, F1DIM, L2, 1);
    fc2_softmax_kernel<<<g1d(RS, TB), TB, 0, stream>>>(f1, fc2_w, fc2_b, out, RS);
}
